// GatedMultiHeadSelfAttention_71236327571470
// MI455X (gfx1250) — matrix operand order repaired and checked by reading, not compiled, not run on hardware
//
#include <hip/hip_runtime.h>
#include <hip/hip_bf16.h>

// GatedMultiHeadSelfAttention, MI455X (gfx1250) bf16-WMMA implementation.
// B=2, T=2048, E=1024, H=16, D=64.
// - All matrix math: v_wmma_f32_16x16x32_bf16 (f32 accumulation)
// - Attention K/V streamed via global_load_async_to_lds_b128 (ASYNCcnt),
//   double-buffered in LDS, shared by 4 waves per workgroup.

#define E_ 1024
#define H_ 16
#define D_ 64
#define B_ 2
#define T_ 2048
#define HD_ (H_ * D_)   // 1024

typedef __attribute__((ext_vector_type(16))) __bf16 v16bf;
typedef __attribute__((ext_vector_type(8)))  float  v8f;

union FragAB { uint4 q[2]; v16bf v; };

__device__ __forceinline__ unsigned short f2bf_bits(float f) {
  unsigned int u = __float_as_uint(f);
  u += 0x7FFFu + ((u >> 16) & 1u);          // round-to-nearest-even
  return (unsigned short)(u >> 16);
}

// A fragment (16x32 bf16, row-major source, stride ld elements).
// Lanes 0-15: row m=lane, K=0..7,16..23 ; lanes 16-31: K=8..15,24..31.
__device__ __forceinline__ v16bf load_frag_a(const unsigned short* base, int ld) {
  const int lane = threadIdx.x & 31;
  const int m    = lane & 15;
  const int k0   = (lane < 16) ? 0 : 8;
  const unsigned short* p = base + (size_t)m * ld + k0;
  FragAB f;
  f.q[0] = *(const uint4*)(p);
  f.q[1] = *(const uint4*)(p + 16);
  return f.v;
}

// B fragment (32x16 bf16) from B^T stored row-major [N, K], stride ld.
// Lanes 0-15: col n=lane, K=0..7,16..23 ; lanes 16-31: K=8..15,24..31.
__device__ __forceinline__ v16bf load_frag_bt(const unsigned short* base, int ld) {
  const int lane = threadIdx.x & 31;
  const int n    = lane & 15;
  const int k0   = (lane < 16) ? 0 : 8;
  const unsigned short* p = base + (size_t)n * ld + k0;
  FragAB f;
  f.q[0] = *(const uint4*)(p);
  f.q[1] = *(const uint4*)(p + 16);
  return f.v;
}

__device__ __forceinline__ v8f wmma_bf16(v16bf a, v16bf b, v8f c) {
  return __builtin_amdgcn_wmma_f32_16x16x32_bf16(false, a, false, b, (short)0, c,
                                                 false, false);
}

// Async copy 16B/lane from global to LDS (ASYNCcnt-tracked, CDNA5).
__device__ __forceinline__ void async_ld_b128(unsigned lds_byte_off, const void* gaddr) {
  asm volatile("global_load_async_to_lds_b128 %0, %1, off"
               :: "v"(lds_byte_off), "v"((unsigned long long)(size_t)gaddr)
               : "memory");
}
__device__ __forceinline__ void wait_asynccnt_8() {
  asm volatile("s_wait_asynccnt 0x8" ::: "memory");
}
__device__ __forceinline__ void wait_asynccnt_0() {
  asm volatile("s_wait_asynccnt 0x0" ::: "memory");
}

// ---------------- prep kernels ----------------

__global__ void k_cvt_f32_bf16(const float* __restrict__ in,
                               unsigned short* __restrict__ out, int n) {
  int i = blockIdx.x * blockDim.x + threadIdx.x;
  if (i < n) out[i] = f2bf_bits(in[i]);
}

// W: [H, E, D] fp32 -> WT: [H, D, E] bf16
__global__ void k_transpose_qkvw(const float* __restrict__ W,
                                 unsigned short* __restrict__ WT) {
  int i = blockIdx.x * blockDim.x + threadIdx.x;   // over H*D*E (output-linear)
  int e = i % E_;
  int r = i / E_;
  int d = r % D_;
  int h = r / D_;
  WT[i] = f2bf_bits(W[((size_t)h * E_ + e) * D_ + d]);
}

// Wo: [H*D, E] fp32 -> WoT: [E, H*D] bf16
__global__ void k_transpose_wo(const float* __restrict__ Wo,
                               unsigned short* __restrict__ WoT) {
  int i  = blockIdx.x * blockDim.x + threadIdx.x;  // over E*H*D (output-linear)
  int hd = i % HD_;
  int e  = i / HD_;
  WoT[i] = f2bf_bits(Wo[(size_t)hd * E_ + e]);
}

// bsum[e] = sum_h gate[h] * bo[h, e]
__global__ void k_bsum(const float* __restrict__ bo, const float* __restrict__ gate,
                       float* __restrict__ bsum) {
  int e = blockIdx.x * blockDim.x + threadIdx.x;
  if (e < E_) {
    float s = 0.f;
    #pragma unroll
    for (int h = 0; h < H_; ++h) s += gate[h] * bo[(size_t)h * E_ + e];
    bsum[e] = s;
  }
}

// ---------------- QKV projection ----------------
// One wave: 32 tokens x 64 (=D) for one (b,h); two accumulator chains share
// every B fragment. Q,K stored [B,H,T,D]; V stored transposed [B,H,D,T].
__global__ void k_qkv(const unsigned short* __restrict__ hsb,
                      const unsigned short* __restrict__ WqT,
                      const unsigned short* __restrict__ WkT,
                      const unsigned short* __restrict__ WvT,
                      const float* __restrict__ bq, const float* __restrict__ bk,
                      const float* __restrict__ bv,
                      unsigned short* __restrict__ Qb,
                      unsigned short* __restrict__ Kb,
                      unsigned short* __restrict__ Vt) {
  const int which = blockIdx.y;                // 0=Q 1=K 2=V
  const int idx   = blockIdx.x;                // b*H*64 + h*64 + mt
  const int mt    = idx & 63;
  const int h     = (idx >> 6) & 15;
  const int b     = idx >> 10;
  const int lane  = threadIdx.x & 31;

  const unsigned short* WT   = (which == 0) ? WqT : (which == 1) ? WkT : WvT;
  const float*          bias = (which == 0) ? bq  : (which == 1) ? bk  : bv;

  const unsigned short* A0 = hsb + ((size_t)b * T_ + (size_t)mt * 32) * E_;
  const unsigned short* A1 = A0 + (size_t)16 * E_;
  const unsigned short* Bt = WT + (size_t)h * D_ * E_;   // [D, E]

  v8f acc0[4] = {}, acc1[4] = {};
  for (int k = 0; k < E_; k += 32) {
    v16bf a0 = load_frag_a(A0 + k, E_);
    v16bf a1 = load_frag_a(A1 + k, E_);
    #pragma unroll
    for (int dt = 0; dt < 4; ++dt) {
      v16bf bb = load_frag_bt(Bt + (size_t)(dt * 16) * E_ + k, E_);
      acc0[dt] = wmma_bf16(a0, bb, acc0[dt]);
      acc1[dt] = wmma_bf16(a1, bb, acc1[dt]);
    }
  }

  const int n     = lane & 15;
  const int mbase = (lane < 16) ? 0 : 8;
  const int t0    = mt * 32;
  #pragma unroll
  for (int half = 0; half < 2; ++half) {
    v8f* acc = half ? acc1 : acc0;
    #pragma unroll
    for (int dt = 0; dt < 4; ++dt) {
      const int   d  = dt * 16 + n;
      const float bb = bias[(size_t)h * D_ + d];
      #pragma unroll
      for (int v = 0; v < 8; ++v) {
        const int t = t0 + half * 16 + mbase + v;
        unsigned short val = f2bf_bits(acc[dt][v] + bb);
        if (which == 2) {
          Vt[(((size_t)b * H_ + h) * D_ + d) * T_ + t] = val;
        } else {
          unsigned short* O = (which == 0) ? Qb : Kb;
          O[(((size_t)b * H_ + h) * T_ + t) * D_ + d] = val;
        }
      }
    }
  }
}

// ---------------- flash attention ----------------
// Block = 4 waves = 64 query rows of one (b,h); wave w owns rows [16w,16w+16).
// KV streamed in 64-column chunks via async-to-LDS, double-buffered; each wave
// stages 1/4 of each chunk (8 async b128 ops) and waits on its own ASYNCcnt,
// then __syncthreads() publishes the buffer block-wide.
__global__ void __launch_bounds__(128)
k_attn(const unsigned short* __restrict__ Qb,
       const unsigned short* __restrict__ Kb,
       const unsigned short* __restrict__ Vt,
       const float* __restrict__ gate,
       unsigned short* __restrict__ ctxb) {
  __shared__ __attribute__((aligned(16))) unsigned short kbuf[2][64 * 64]; // [tok][d]
  __shared__ __attribute__((aligned(16))) unsigned short vbuf[2][64 * 64]; // [d][tok]
  __shared__ __attribute__((aligned(16))) unsigned short pls[4][16 * 64];  // per-wave P

  const int idx    = blockIdx.x;               // b*H*32 + h*32 + ct
  const int ct     = idx & 31;                 // 64-row query chunk
  const int h      = (idx >> 5) & 15;
  const int b      = idx >> 9;
  const int lane   = threadIdx.x & 31;
  const int waveId = threadIdx.x >> 5;
  const int t0     = ct * 64 + waveId * 16;

  const unsigned short* Qbase = Qb + (((size_t)b * H_ + h) * T_ + t0) * D_;
  const unsigned short* Kbase = Kb + (((size_t)b * H_ + h) * T_) * D_;   // [T,D]
  const unsigned short* Vbase = Vt + (((size_t)b * H_ + h) * D_) * T_;   // [D,T]

  const v16bf aq0 = load_frag_a(Qbase, D_);        // K = d 0..31
  const v16bf aq1 = load_frag_a(Qbase + 32, D_);   // K = d 32..63

  v8f acc[4] = {};
  float mrow[8], lrow[8];
  #pragma unroll
  for (int v = 0; v < 8; ++v) { mrow[v] = -1e30f; lrow[v] = 0.f; }

  const float scale = 0.125f;                  // 1/sqrt(D)
  const int n      = lane & 15;
  const int mbase  = (lane < 16) ? 0 : 8;
  unsigned short* pw = &pls[waveId][0];

  // Stage one 64-column chunk quarter (this wave's share: 8 async b128 ops).
  auto stage = [&](int s0, int buf) {
    const unsigned kb_off = (unsigned)(size_t)&kbuf[buf][0];
    const unsigned vb_off = (unsigned)(size_t)&vbuf[buf][0];
    const char* Kg = (const char*)(Kbase + (size_t)s0 * D_);     // 8KB contiguous
    const char* Vg = (const char*)(Vbase + s0);                  // 64 rows x 128B, stride T*2
    #pragma unroll
    for (int i = 0; i < 4; ++i) {
      unsigned lin = (unsigned)(waveId * 2048 + i * 512 + lane * 16);
      async_ld_b128(kb_off + lin, Kg + lin);
    }
    #pragma unroll
    for (int i = 0; i < 4; ++i) {
      unsigned lin = (unsigned)(waveId * 2048 + i * 512 + lane * 16);
      unsigned row = lin >> 7, byte = lin & 127;
      async_ld_b128(vb_off + lin, Vg + (size_t)row * (T_ * 2) + byte);
    }
  };

  const int NCHUNK = T_ / 64;                  // 32
  stage(0, 0);
  for (int c = 0; c < NCHUNK; ++c) {
    if (c + 1 < NCHUNK) { stage((c + 1) * 64, (c + 1) & 1); wait_asynccnt_8(); }
    else               { wait_asynccnt_0(); }
    __syncthreads();                            // chunk c visible block-wide

    const unsigned short* kc = &kbuf[c & 1][0]; // [64 tok][64 d]
    const unsigned short* vc = &vbuf[c & 1][0]; // [64 d][64 tok]

    // S tile 16x64 = four 16x16 accumulators (8 WMMA)
    v8f s[4];
    #pragma unroll
    for (int j = 0; j < 4; ++j) {
      v8f sj = {};
      sj   = wmma_bf16(aq0, load_frag_bt(kc + (size_t)(j * 16) * 64, 64), sj);
      sj   = wmma_bf16(aq1, load_frag_bt(kc + (size_t)(j * 16) * 64 + 32, 64), sj);
      s[j] = sj;
    }

    // online softmax; row stats identical across each 16-lane group
    #pragma unroll
    for (int v = 0; v < 8; ++v) {
      float a0 = s[0][v] * scale, a1 = s[1][v] * scale;
      float a2 = s[2][v] * scale, a3 = s[3][v] * scale;
      float rm = fmaxf(fmaxf(a0, a1), fmaxf(a2, a3));
      rm = fmaxf(rm, __shfl_xor(rm, 1, 32));
      rm = fmaxf(rm, __shfl_xor(rm, 2, 32));
      rm = fmaxf(rm, __shfl_xor(rm, 4, 32));
      rm = fmaxf(rm, __shfl_xor(rm, 8, 32));
      float mnew = fmaxf(mrow[v], rm);
      float corr = __expf(mrow[v] - mnew);
      float p0 = __expf(a0 - mnew), p1 = __expf(a1 - mnew);
      float p2 = __expf(a2 - mnew), p3 = __expf(a3 - mnew);
      float rs = (p0 + p1) + (p2 + p3);
      rs += __shfl_xor(rs, 1, 32);
      rs += __shfl_xor(rs, 2, 32);
      rs += __shfl_xor(rs, 4, 32);
      rs += __shfl_xor(rs, 8, 32);
      lrow[v] = lrow[v] * corr + rs;
      mrow[v] = mnew;
      #pragma unroll
      for (int dt = 0; dt < 4; ++dt) acc[dt][v] *= corr;
      pw[(mbase + v) * 64 + n]      = f2bf_bits(p0);
      pw[(mbase + v) * 64 + 16 + n] = f2bf_bits(p1);
      pw[(mbase + v) * 64 + 32 + n] = f2bf_bits(p2);
      pw[(mbase + v) * 64 + 48 + n] = f2bf_bits(p3);
    }
    asm volatile("" ::: "memory");   // per-wave LDS in-order: stores before frag loads

    // P (16x64) as two A fragments; P*V: 8 WMMA
    v16bf ap0 = load_frag_a(pw, 64);
    v16bf ap1 = load_frag_a(pw + 32, 64);
    #pragma unroll
    for (int dt = 0; dt < 4; ++dt) {
      acc[dt] = wmma_bf16(ap0, load_frag_bt(vc + (size_t)(dt * 16) * 64, 64), acc[dt]);
      acc[dt] = wmma_bf16(ap1, load_frag_bt(vc + (size_t)(dt * 16) * 64 + 32, 64), acc[dt]);
    }
    __syncthreads();                            // all reads done before buffer reuse
  }

  // normalize, fold gate, store ctx bf16 as [B, T, H*D]
  const float g = gate[h];
  unsigned short* Obase = ctxb + ((size_t)b * T_ + t0) * HD_ + (size_t)h * D_;
  #pragma unroll
  for (int dt = 0; dt < 4; ++dt) {
    const int d = dt * 16 + n;
    #pragma unroll
    for (int v = 0; v < 8; ++v) {
      float val = acc[dt][v] / lrow[v] * g;
      Obase[(size_t)(mbase + v) * HD_ + d] = f2bf_bits(val);
    }
  }
}

// ---------------- output projection ----------------
// One wave: 32 tokens x 64 e-columns; two accumulator chains share B fragments.
__global__ void k_oproj(const unsigned short* __restrict__ ctxb,
                        const unsigned short* __restrict__ WoT,
                        const float* __restrict__ bsum,
                        float* __restrict__ out) {
  const int idx  = blockIdx.x;                 // b*64*16 + mt*16 + nt
  const int nt   = idx & 15;
  const int mt   = (idx >> 4) & 63;
  const int b    = idx >> 10;
  const int lane = threadIdx.x & 31;
  const int t0   = mt * 32;
  const int e0   = nt * 64;

  const unsigned short* A0 = ctxb + ((size_t)b * T_ + t0) * HD_;
  const unsigned short* A1 = A0 + (size_t)16 * HD_;

  v8f acc0[4] = {}, acc1[4] = {};
  for (int k = 0; k < HD_; k += 32) {
    v16bf a0 = load_frag_a(A0 + k, HD_);
    v16bf a1 = load_frag_a(A1 + k, HD_);
    #pragma unroll
    for (int et = 0; et < 4; ++et) {
      v16bf bb = load_frag_bt(WoT + (size_t)(e0 + et * 16) * HD_ + k, HD_);
      acc0[et] = wmma_bf16(a0, bb, acc0[et]);
      acc1[et] = wmma_bf16(a1, bb, acc1[et]);
    }
  }

  const int n     = lane & 15;
  const int mbase = (lane < 16) ? 0 : 8;
  float* Obase = out + ((size_t)b * T_ + t0) * E_;
  #pragma unroll
  for (int half = 0; half < 2; ++half) {
    v8f* acc = half ? acc1 : acc0;
    #pragma unroll
    for (int et = 0; et < 4; ++et) {
      const int   e  = e0 + et * 16 + n;
      const float bs = bsum[e];
      #pragma unroll
      for (int v = 0; v < 8; ++v) {
        Obase[(size_t)(half * 16 + mbase + v) * E_ + e] = acc[et][v] + bs;
      }
    }
  }
}

// ---------------- host launcher ----------------
extern "C" void kernel_launch(void* const* d_in, const int* in_sizes, int n_in,
                              void* d_out, int out_size, void* d_ws, size_t ws_size,
                              hipStream_t stream) {
  (void)in_sizes; (void)n_in; (void)out_size; (void)ws_size;
  const float* hs   = (const float*)d_in[0];
  const float* Wq   = (const float*)d_in[1];
  const float* bq   = (const float*)d_in[2];
  const float* Wk   = (const float*)d_in[3];
  const float* bk   = (const float*)d_in[4];
  const float* Wv   = (const float*)d_in[5];
  const float* bv   = (const float*)d_in[6];
  const float* Wo   = (const float*)d_in[7];
  const float* bo   = (const float*)d_in[8];
  const float* gate = (const float*)d_in[9];
  float* out = (float*)d_out;

  char*  ws  = (char*)d_ws;
  size_t off = 0;
  auto alloc = [&](size_t bytes) -> char* {
    char* p = ws + off;
    off += (bytes + 255) & ~(size_t)255;
    return p;
  };
  unsigned short* hsb  = (unsigned short*)alloc((size_t)B_ * T_ * E_ * 2);
  unsigned short* WqT  = (unsigned short*)alloc((size_t)H_ * D_ * E_ * 2);
  unsigned short* WkT  = (unsigned short*)alloc((size_t)H_ * D_ * E_ * 2);
  unsigned short* WvT  = (unsigned short*)alloc((size_t)H_ * D_ * E_ * 2);
  unsigned short* WoT  = (unsigned short*)alloc((size_t)E_ * HD_ * 2);
  unsigned short* Qb   = (unsigned short*)alloc((size_t)B_ * H_ * T_ * D_ * 2);
  unsigned short* Kb   = (unsigned short*)alloc((size_t)B_ * H_ * T_ * D_ * 2);
  unsigned short* Vt   = (unsigned short*)alloc((size_t)B_ * H_ * T_ * D_ * 2);
  unsigned short* ctxb = (unsigned short*)alloc((size_t)B_ * T_ * HD_ * 2);
  float*          bsum = (float*)alloc((size_t)E_ * 4);

  // prep
  k_cvt_f32_bf16<<<(B_ * T_ * E_) / 256, 256, 0, stream>>>(hs, hsb, B_ * T_ * E_);
  k_transpose_qkvw<<<(H_ * D_ * E_) / 256, 256, 0, stream>>>(Wq, WqT);
  k_transpose_qkvw<<<(H_ * D_ * E_) / 256, 256, 0, stream>>>(Wk, WkT);
  k_transpose_qkvw<<<(H_ * D_ * E_) / 256, 256, 0, stream>>>(Wv, WvT);
  k_transpose_wo<<<(E_ * HD_) / 256, 256, 0, stream>>>(Wo, WoT);
  k_bsum<<<E_ / 256, 256, 0, stream>>>(bo, gate, bsum);

  // QKV projections (grid.y selects Q/K/V), 32-row tiles
  k_qkv<<<dim3(B_ * H_ * (T_ / 32), 3), 32, 0, stream>>>(hsb, WqT, WkT, WvT,
                                                         bq, bk, bv, Qb, Kb, Vt);
  // flash attention: 4-wave blocks, async-to-LDS double-buffered KV
  k_attn<<<B_ * H_ * (T_ / 64), 128, 0, stream>>>(Qb, Kb, Vt, gate, ctxb);
  // output projection + gated-bias epilogue, 32-row tiles
  k_oproj<<<B_ * (T_ / 32) * (E_ / 64), 32, 0, stream>>>(ctxb, WoT, bsum, out);
}
